// MoELayer_66116726555021
// MI455X (gfx1250) — matrix operand order repaired and checked by reading, not compiled, not run on hardware
//
#include <hip/hip_runtime.h>
#include <hip/hip_bf16.h>

// ---------------------------------------------------------------------------
// MoE top-1 routed FFN for MI455X (gfx1250, wave32, WMMA bf16 16x16x32).
// Sizes fixed by the reference: B=4, L=2048, D=1024, E=8, H=4096, N=8192.
// Top-1 routing => compute only the selected expert per token (1/8 the dense
// FLOPs, ~137 GFLOP).  Heavy GEMMs in bf16 WMMA w/ f32 accumulate; gate in
// exact fp32 so the argmax selection matches the reference.
// ---------------------------------------------------------------------------

#define NTOK 8192
#define DDIM 1024
#define EEXP 8
#define HDIM 4096

typedef __bf16 bf16_t;
typedef __attribute__((ext_vector_type(16))) __bf16 v16bf;
typedef __attribute__((ext_vector_type(8)))  __bf16 v8bf;
typedef __attribute__((ext_vector_type(4)))  __bf16 v4bf;
typedef __attribute__((ext_vector_type(2)))  __bf16 v2bf;
typedef __attribute__((ext_vector_type(8)))  float  v8f;

// ---------------------------------------------------------------------------
// Gate: fp32 exact.  One wave (32 lanes) per token; 8 tokens per block.
// ---------------------------------------------------------------------------
__global__ __launch_bounds__(256)
void gate_kernel(const float* __restrict__ x, const float* __restrict__ Wg,
                 const float* __restrict__ bg, float* __restrict__ gate_out,
                 int* __restrict__ top_idx, float* __restrict__ top_w) {
  const int lane = threadIdx.x & 31;
  const int wid  = threadIdx.x >> 5;
  const int n = blockIdx.x * 8 + wid;
  if (n >= NTOK) return;
  const float* xr = x + (size_t)n * DDIM;
  float acc[EEXP];
#pragma unroll
  for (int e = 0; e < EEXP; ++e) acc[e] = 0.f;
  for (int d = lane; d < DDIM; d += 32) {
    const float xv = xr[d];
    const float* wr = Wg + (size_t)d * EEXP;
#pragma unroll
    for (int e = 0; e < EEXP; ++e) acc[e] += xv * wr[e];
  }
#pragma unroll
  for (int e = 0; e < EEXP; ++e) {
    float v = acc[e];
    for (int off = 16; off > 0; off >>= 1) v += __shfl_xor(v, off, 32);
    acc[e] = v;
  }
  if (lane == 0) {
    float mx = -3.0e38f;
#pragma unroll
    for (int e = 0; e < EEXP; ++e) { acc[e] += bg[e]; mx = acc[e] > mx ? acc[e] : mx; }
    float w[EEXP], s = 0.f;
#pragma unroll
    for (int e = 0; e < EEXP; ++e) { w[e] = __expf(acc[e] - mx); s += w[e]; }
    const float inv = 1.f / s;
    int   bi = 0;
    float bw = -1.f;
#pragma unroll
    for (int e = 0; e < EEXP; ++e) {
      w[e] *= inv;
      gate_out[(size_t)n * EEXP + e] = w[e];
      if (w[e] > bw) { bw = w[e]; bi = e; }   // strict > => lowest index on tie
    }
    top_idx[n] = bi;
    top_w[n]  = bw;
  }
}

__global__ void zero_cnt(int* __restrict__ cnt) {
  if (threadIdx.x < 64) cnt[threadIdx.x] = 0;
}

__global__ __launch_bounds__(256)
void assign_kernel(const int* __restrict__ top_idx, int* __restrict__ cnt,
                   int* __restrict__ list) {
  const int n = blockIdx.x * 256 + threadIdx.x;
  if (n >= NTOK) return;
  const int e = top_idx[n];
  const int pos = atomicAdd(&cnt[e], 1);
  list[e * NTOK + pos] = n;
}

__global__ __launch_bounds__(256)
void cvt_x(const float* __restrict__ x, bf16_t* __restrict__ xb) {
  const size_t i = ((size_t)blockIdx.x * 256 + threadIdx.x) * 4;
  const float4 v = *(const float4*)(x + i);
  v4bf o;
  o[0] = (bf16_t)v.x; o[1] = (bf16_t)v.y; o[2] = (bf16_t)v.z; o[3] = (bf16_t)v.w;
  *(v4bf*)(xb + i) = o;
}

// ---------------------------------------------------------------------------
// Routed GEMM, per expert.  Workgroup tile 128(M) x 128(N), K-step 32.
// 8 waves; wave tile 64x32 = 4x2 WMMA 16x16 accumulators (f32).
// Software-pipelined: next K-step's global loads are issued into registers
// before the WMMA block, hiding HBM/L2 latency behind matrix math.
// STAGE 1:  h = relu(X_bf16 @ W1[e] + b1[e])        -> Hbf (bf16, per token)
// STAGE 2:  out[tok] = top_w * (h @ W2[e] + b2[e])  -> f32 scatter
// ---------------------------------------------------------------------------
template <int STAGE>
__global__ __launch_bounds__(256)
void ffn_gemm(const bf16_t* __restrict__ Abase,  // row stride = KDIM (per token)
              const float*  __restrict__ Wall,   // [E, KDIM, NDIM] fp32
              const float*  __restrict__ bias,   // [E, NDIM]
              const int*    __restrict__ cnt,
              const int*    __restrict__ list,   // [E, NTOK]
              const float*  __restrict__ top_w,
              bf16_t* __restrict__ Hout,         // stage1 output, row stride HDIM
              float*  __restrict__ Fout) {       // stage2 output, row stride DDIM
  constexpr int KDIM = (STAGE == 1) ? DDIM : HDIM;
  constexpr int NDIM = (STAGE == 1) ? HDIM : DDIM;

  const int e = blockIdx.z;
  const int count = cnt[e];
  const int m0 = blockIdx.x * 128;
  if (m0 >= count) return;
  const int n0 = blockIdx.y * 128;

  const int* mylist = list + e * NTOK;
  const float* We = Wall + (size_t)e * KDIM * NDIM;

  __shared__ bf16_t lA[128][32];   // [m][k]
  __shared__ bf16_t lB[128][32];   // [n][k]  (transposed while staging)

  const int tid  = threadIdx.x;
  const int lane = tid & 31;
  const int wid  = tid >> 5;
  const int wm = (wid & 1) * 64;   // wave M base in tile
  const int wn = (wid >> 1) * 32;  // wave N base in tile

  // --- A staging: thread -> (row, k-half); rows are gathered token rows ---
  const int ar = tid >> 1;               // 0..127
  const int ah = (tid & 1) * 16;         // k offset {0,16}
  const int apos = m0 + ar;
  const int aidx = apos < count ? apos : (count - 1);
  const bf16_t* arow = Abase + (size_t)mylist[aidx] * KDIM;

  // --- B staging: thread -> (2 k-rows, 8 n) with packed v2bf stores ---
  const int bkr = (tid >> 4) * 2;        // 0,2,...,30
  const int bns = (tid & 15) * 8;        // 0..120

  v8f acc[4][2];
#pragma unroll
  for (int i = 0; i < 4; ++i)
#pragma unroll
    for (int j = 0; j < 2; ++j)
#pragma unroll
      for (int q = 0; q < 8; ++q) acc[i][j][q] = 0.f;

  const int c0 = (lane < 16) ? 0 : 8;    // A fragment K base (16-bit A 16x32 layout)
  const int kb = (lane < 16) ? 0 : 16;   // unused; B fragment reads use c0, same k order as A
  const int lrow = lane & 15;

  // Register prefetch buffers (next K-step's global data)
  v8bf   pa0, pa1;
  float4 pb0, pb1, pb2, pb3;

  auto load_global = [&](int k0) {
    const v8bf* src = (const v8bf*)(arow + k0 + ah);
    pa0 = src[0];
    pa1 = src[1];
    const float* b0 = We + (size_t)(k0 + bkr) * NDIM + (n0 + bns);
    pb0 = *(const float4*)(b0);
    pb1 = *(const float4*)(b0 + 4);
    pb2 = *(const float4*)(b0 + NDIM);
    pb3 = *(const float4*)(b0 + NDIM + 4);
  };

  load_global(0);

  for (int k0 = 0; k0 < KDIM; k0 += 32) {
    __syncthreads();   // previous iteration's fragment reads done
    // ---- commit prefetched tile to LDS ----
    *(v8bf*)(&lA[ar][ah])     = pa0;
    *(v8bf*)(&lA[ar][ah + 8]) = pa1;
    {
      const float r0[8] = {pb0.x, pb0.y, pb0.z, pb0.w, pb1.x, pb1.y, pb1.z, pb1.w};
      const float r1[8] = {pb2.x, pb2.y, pb2.z, pb2.w, pb3.x, pb3.y, pb3.z, pb3.w};
#pragma unroll
      for (int q = 0; q < 8; ++q) {
        v2bf p;
        p[0] = (bf16_t)r0[q];
        p[1] = (bf16_t)r1[q];
        *(v2bf*)(&lB[bns + q][bkr]) = p;   // packed 32-bit DS store
      }
    }
    __syncthreads();

    // ---- issue next K-step's global loads (overlaps with WMMA below) ----
    if (k0 + 32 < KDIM) load_global(k0 + 32);

    // ---- fragments + WMMA ----
    v16bf afr[4];
#pragma unroll
    for (int i = 0; i < 4; ++i) {
      const int m = wm + i * 16 + lrow;
      *(v8bf*)(&afr[i])       = *(const v8bf*)(&lA[m][c0]);        // K = c0..c0+7
      *((v8bf*)(&afr[i]) + 1) = *(const v8bf*)(&lA[m][c0 + 16]);   // K = c0+16..c0+23
    }
    v16bf bfr[2];
#pragma unroll
    for (int j = 0; j < 2; ++j) {
      const int n = wn + j * 16 + lrow;
      *(v8bf*)(&bfr[j])       = *(const v8bf*)(&lB[n][c0]);
      *((v8bf*)(&bfr[j]) + 1) = *(const v8bf*)(&lB[n][c0 + 16]);
    }
#pragma unroll
    for (int i = 0; i < 4; ++i)
#pragma unroll
      for (int j = 0; j < 2; ++j)
        acc[i][j] = __builtin_amdgcn_wmma_f32_16x16x32_bf16(
            false, afr[i], false, bfr[j], (short)0, acc[i][j], false, false);
  }

  // --- epilogue: C/D layout: VGPR v -> M = v (+8 for lanes 16..31), N = lane%16
  const int lm = (lane < 16) ? 0 : 8;
  float bv[2];
#pragma unroll
  for (int j = 0; j < 2; ++j)
    bv[j] = bias[(size_t)e * NDIM + n0 + wn + j * 16 + lrow];

#pragma unroll
  for (int i = 0; i < 4; ++i) {
#pragma unroll
    for (int v = 0; v < 8; ++v) {
      const int pos = m0 + wm + i * 16 + lm + v;
      if (pos < count) {
        const int tok = mylist[pos];               // one row lookup per output row
        if constexpr (STAGE == 1) {
          bf16_t* orow = Hout + (size_t)tok * HDIM + n0 + wn + lrow;
#pragma unroll
          for (int j = 0; j < 2; ++j) {
            const float val = acc[i][j][v] + bv[j];
            orow[j * 16] = (bf16_t)(val > 0.f ? val : 0.f);
          }
        } else {
          const float tw = top_w[tok];
          float* orow = Fout + (size_t)tok * DDIM + n0 + wn + lrow;
#pragma unroll
          for (int j = 0; j < 2; ++j)
            orow[j * 16] = tw * (acc[i][j][v] + bv[j]);
        }
      }
    }
  }
}

// ---------------------------------------------------------------------------
// Host launcher
// ---------------------------------------------------------------------------
extern "C" void kernel_launch(void* const* d_in, const int* in_sizes, int n_in,
                              void* d_out, int out_size, void* d_ws, size_t ws_size,
                              hipStream_t stream) {
  (void)in_sizes; (void)n_in; (void)out_size; (void)ws_size;
  const float* x  = (const float*)d_in[0];
  const float* W1 = (const float*)d_in[1];
  const float* b1 = (const float*)d_in[2];
  const float* W2 = (const float*)d_in[3];
  const float* b2 = (const float*)d_in[4];
  const float* Wg = (const float*)d_in[5];
  const float* bg = (const float*)d_in[6];

  float* out      = (float*)d_out;                    // [N, D]
  float* gate_out = out + (size_t)NTOK * DDIM;        // [N, E]

  char* w = (char*)d_ws;
  int*    top_idx = (int*)w;     w += (size_t)NTOK * sizeof(int);
  float*  top_w   = (float*)w;   w += (size_t)NTOK * sizeof(float);
  int*    cnt     = (int*)w;     w += 256;
  int*    list    = (int*)w;     w += (size_t)EEXP * NTOK * sizeof(int);
  bf16_t* Xbf     = (bf16_t*)w;  w += (size_t)NTOK * DDIM * sizeof(bf16_t);
  bf16_t* Hbf     = (bf16_t*)w;  // NTOK * HDIM bf16

  zero_cnt<<<1, 64, 0, stream>>>(cnt);
  gate_kernel<<<NTOK / 8, 256, 0, stream>>>(x, Wg, bg, gate_out, top_idx, top_w);
  assign_kernel<<<NTOK / 256, 256, 0, stream>>>(top_idx, cnt, list);
  cvt_x<<<(NTOK * DDIM) / 1024, 256, 0, stream>>>(x, Xbf);

  dim3 g1(NTOK / 128, HDIM / 128, EEXP);
  ffn_gemm<1><<<g1, 256, 0, stream>>>(Xbf, W1, b1, cnt, list, top_w, Hbf, nullptr);
  dim3 g2(NTOK / 128, DDIM / 128, EEXP);
  ffn_gemm<2><<<g2, 256, 0, stream>>>(Hbf, W2, b2, cnt, list, top_w, nullptr, out);
}
